// SelfAttention_60619168415788
// MI455X (gfx1250) — matrix operand order repaired and checked by reading, not compiled, not run on hardware
//
#include <hip/hip_runtime.h>

typedef __attribute__((ext_vector_type(16))) __bf16 v16bf;
typedef __attribute__((ext_vector_type(8)))  __bf16 v8bf;
typedef __attribute__((ext_vector_type(8)))  float  v8f;
typedef __attribute__((ext_vector_type(4)))  int    v4i;
typedef __attribute__((ext_vector_type(8)))  int    v8i;

static constexpr int D_EMBED = 1024;
static constexpr int N_HEADS = 16;
static constexpr int D_HEAD  = 64;
static constexpr int QKV_W   = 3 * D_EMBED;   // 3072
static constexpr int SEQ     = 2048;
static constexpr int BATCH   = 2;
static constexpr int BS      = BATCH * SEQ;   // 4096

__device__ __forceinline__ v8f wmma_bf16(v16bf a, v16bf b, v8f c) {
  // D = A(16x32 bf16) * B(32x16 bf16) + C(16x16 f32)
  return __builtin_amdgcn_wmma_f32_16x16x32_bf16(false, a, false, b, (short)0, c,
                                                 false, false);
}

__device__ __forceinline__ v16bf join8(v8bf lo, v8bf hi) {
  return __builtin_shufflevector(lo, hi, 0, 1, 2, 3, 4, 5, 6, 7,
                                 8, 9, 10, 11, 12, 13, 14, 15);
}

__device__ __forceinline__ v8f zero8f() {
  v8f z = {0.f, 0.f, 0.f, 0.f, 0.f, 0.f, 0.f, 0.f};
  return z;
}

// ------------------------------------------------------------------ TDM
// Issue a 2D tile load (tile_w x tile_h bf16, row stride = stride_elems)
// from global memory into LDS at byte offset lds_off via the Tensor Data
// Mover. D# layout per CDNA5 ISA ch.8: group0 (count/type/addresses),
// group1 (data_size, dims, tile dims, strides). VADDR2/3 = NULL (2D).
__device__ __forceinline__ void tdm_load_tile(const __bf16* gsrc,
                                              unsigned lds_off,
                                              unsigned tile_w, unsigned tile_h,
                                              unsigned stride_elems) {
  unsigned long long ga = (unsigned long long)(uintptr_t)gsrc;
  const unsigned td0 = 0x100000u, td1 = 0x100000u;  // huge tensor dims: no OOB clip
  v4i g0;
  g0[0] = 1;                                            // count=1, user descriptor
  g0[1] = (int)lds_off;                                 // lds_addr (bytes)
  g0[2] = (int)(unsigned)(ga & 0xffffffffull);          // global_addr[31:0]
  g0[3] = (int)((unsigned)((ga >> 32) & 0x01ffffffull)  // global_addr[56:32]
                | 0x80000000u);                         // type=2 ("image")
  v8i g1;
  g1[0] = (int)(1u << 16);                              // data_size=1 -> 2 bytes
  g1[1] = (int)((td0 & 0xffffu) << 16);                 // tensor_dim0[15:0]
  g1[2] = (int)(((td0 >> 16) & 0xffffu) | ((td1 & 0xffffu) << 16));
  g1[3] = (int)(((td1 >> 16) & 0xffffu) | ((tile_w & 0xffffu) << 16)); // tile_dim0
  g1[4] = (int)(tile_h & 0xffffu);                      // tile_dim1, tile_dim2=0
  g1[5] = (int)stride_elems;                            // tensor_dim0_stride[31:0]
  g1[6] = 0;                                            // stride hi, dim1_stride lo
  g1[7] = 0;
  asm volatile("tensor_load_to_lds %0, %1" :: "s"(g0), "s"(g1) : "memory");
}

// ---------------------------------------------------------------- conversion
__global__ void cvt_f32_bf16_kernel(const float* __restrict__ x,
                                    __bf16* __restrict__ y, int n) {
  int i = blockIdx.x * blockDim.x + threadIdx.x;
  if (i < n) y[i] = (__bf16)x[i];
}

// Vt[bh][d][s] = qkv[b*S + s][2*D + h*64 + d]
__global__ void build_vt_kernel(const __bf16* __restrict__ qkv,
                                __bf16* __restrict__ vt) {
  int idx = blockIdx.x * blockDim.x + threadIdx.x;   // over B*H*Dh*S = 4M
  int s  = idx & (SEQ - 1);
  int t  = idx >> 11;              // /SEQ
  int d  = t & (D_HEAD - 1);
  int bh = t >> 6;
  int b  = bh >> 4;
  int h  = bh & 15;
  vt[idx] = qkv[(size_t)(b * SEQ + s) * QKV_W + 2 * D_EMBED + h * D_HEAD + d];
}

// ---------------------------------------------------------------- GEMM
// C[M,N] = A[M,K] * B[N,K]^T + bias[N]   (A,B bf16 row-major, K-contiguous)
// block: 256 threads = 8 waves in 2(M) x 4(N); wave tile 32x64; block 64x256.
// TDM double-buffers the 64x32 A tile and 256x32 B tile in LDS; wave 0
// issues descriptors and gates on TENSORcnt; all waves consume via ds_load.
template <typename OutT>
__global__ void __launch_bounds__(256, 1)
gemm_bf16_wmma_kernel(const __bf16* __restrict__ A,
                      const __bf16* __restrict__ Bm,
                      const float* __restrict__ bias,
                      OutT* __restrict__ C,
                      int M, int N, int K) {
  __shared__ __bf16 Abuf[2][64 * 32];    // 2 x 4 KiB
  __shared__ __bf16 Bbuf[2][256 * 32];   // 2 x 16 KiB
  const int lane = threadIdx.x & 31;
  const int w    = threadIdx.x >> 5;       // 0..7
  const int l15  = lane & 15;
  const int lh   = lane >> 4;              // 0/1
  const int wm   = w & 1;
  const int wn   = w >> 1;
  const int mblk = blockIdx.y * 64;
  const int nblk = blockIdx.x * 256;
  const bool issuer = (threadIdx.x < 32);  // wave 0 drives the TDM

  const unsigned aoff0 = (unsigned)(uintptr_t)&Abuf[0][0];
  const unsigned aoff1 = (unsigned)(uintptr_t)&Abuf[1][0];
  const unsigned boff0 = (unsigned)(uintptr_t)&Bbuf[0][0];
  const unsigned boff1 = (unsigned)(uintptr_t)&Bbuf[1][0];

  v8f acc[2][4];
#pragma unroll
  for (int mt = 0; mt < 2; ++mt)
#pragma unroll
    for (int nt = 0; nt < 4; ++nt) acc[mt][nt] = zero8f();

  if (issuer) {  // prime buffer 0 with the k=0 tiles
    tdm_load_tile(A + (size_t)mblk * K, aoff0, 32, 64, (unsigned)K);
    tdm_load_tile(Bm + (size_t)nblk * K, boff0, 32, 256, (unsigned)K);
  }

  for (int k = 0; k < K; k += 32) {
    const int buf = (k >> 5) & 1;
    if (issuer) {
      if (k + 32 < K) {  // async-prefetch next tiles into the other buffer
        tdm_load_tile(A + (size_t)mblk * K + (k + 32), buf ? aoff0 : aoff1,
                      32, 64, (unsigned)K);
        tdm_load_tile(Bm + (size_t)nblk * K + (k + 32), buf ? boff0 : boff1,
                      32, 256, (unsigned)K);
        __builtin_amdgcn_s_wait_tensorcnt(2);  // current pair done, next in flight
      } else {
        __builtin_amdgcn_s_wait_tensorcnt(0);
      }
    }
    __syncthreads();  // current buffer visible to all waves

    v16bf af[2];
#pragma unroll
    for (int mt = 0; mt < 2; ++mt) {
      // A fragment: lane row = wm*32+mt*16+l15; elems 0..15 = K[lh*16..],
      // contiguous
      const __bf16* ap = &Abuf[buf][(wm * 32 + mt * 16 + l15) * 32 + lh * 16];
      af[mt] = join8(*(const v8bf*)ap, *(const v8bf*)(ap + 8));
    }
#pragma unroll
    for (int nt = 0; nt < 4; ++nt) {
      // B fragment: lane col row = wn*64+nt*16+l15; 16 contiguous K at lh*16
      const __bf16* bp = &Bbuf[buf][(wn * 64 + nt * 16 + l15) * 32 + lh * 16];
      v16bf bfr = *(const v16bf*)bp;
#pragma unroll
      for (int mt = 0; mt < 2; ++mt)
        acc[mt][nt] = wmma_bf16(af[mt], bfr, acc[mt][nt]);
    }
    __syncthreads();  // all waves done with buffer before TDM overwrites it
  }

#pragma unroll
  for (int nt = 0; nt < 4; ++nt) {
    const int col = nblk + wn * 64 + nt * 16 + l15;
    const float bv = bias[col];
#pragma unroll
    for (int mt = 0; mt < 2; ++mt)
#pragma unroll
      for (int i = 0; i < 8; ++i) {
        // C layout: VGPR i -> row m = i + 8*lh; col = l15
        const int row = mblk + wm * 32 + mt * 16 + lh * 8 + i;
        C[(size_t)row * N + col] = (OutT)(acc[mt][nt][i] + bv);
      }
  }
}

// ---------------------------------------------------------------- flash attn
// One wave handles 16 query rows for one (b,h). Online softmax over key
// blocks of 32, QK^T and P*V on WMMA, P repacked C-layout -> A-layout via LDS.
// K fragments load before the score WMMAs; V loads issue right after them so
// they overlap the softmax VALU work. Mask VALU only runs on partial blocks.
__global__ void __launch_bounds__(256, 1)
flash_attn_kernel(const __bf16* __restrict__ qkv,
                  const __bf16* __restrict__ vt,
                  __bf16* __restrict__ O,
                  const int* __restrict__ causal_ptr) {
  __shared__ __bf16 pbuf[8][16 * 32];
  const int causal = *causal_ptr;
  const int lane = threadIdx.x & 31;
  const int wid  = threadIdx.x >> 5;
  const int l15  = lane & 15;
  const int lh   = lane >> 4;
  const int gw   = blockIdx.x * 8 + wid;       // 0..4095
  const int qt   = gw & (SEQ / 16 - 1);        // 0..127
  const int bh   = gw >> 7;                    // 0..31
  const int b    = bh >> 4;
  const int h    = bh & 15;
  const int q0   = qt * 16;

  const __bf16* qbase = qkv + (size_t)b * SEQ * QKV_W + h * D_HEAD;
  const __bf16* kbase = qbase + D_EMBED;
  const __bf16* vbase = vt + (size_t)bh * D_HEAD * SEQ;

  // Q fragments (Dh=64 -> two K=32 fragments)
  v16bf qf[2];
#pragma unroll
  for (int kk = 0; kk < 2; ++kk) {
    const __bf16* ap = qbase + (size_t)(q0 + l15) * QKV_W + kk * 32 + lh * 16;
    qf[kk] = join8(*(const v8bf*)ap, *(const v8bf*)(ap + 8));
  }

  v8f oacc[4];
#pragma unroll
  for (int nt = 0; nt < 4; ++nt) oacc[nt] = zero8f();
  float m_run[8], l_run[8];
#pragma unroll
  for (int i = 0; i < 8; ++i) { m_run[i] = -1e30f; l_run[i] = 0.f; }

  const int jmax = causal ? (q0 + 16) : SEQ;
  for (int j = 0; j < jmax; j += 32) {
    // ---- K fragments for this j-block
    v16bf kf[2][2];   // [half][kk]
#pragma unroll
    for (int half = 0; half < 2; ++half)
#pragma unroll
      for (int kk = 0; kk < 2; ++kk) {
        const __bf16* kp = kbase + (size_t)(j + half * 16 + l15) * QKV_W +
                           kk * 32 + lh * 16;
        kf[half][kk] = *(const v16bf*)kp;
      }

    // ---- scores for keys j..j+31
    v8f sc[2];
#pragma unroll
    for (int half = 0; half < 2; ++half) {
      v8f s = zero8f();
      s = wmma_bf16(qf[0], kf[half][0], s);
      s = wmma_bf16(qf[1], kf[half][1], s);
      sc[half] = s;
    }

    // ---- issue V loads now: they complete while we run the softmax VALU
    v16bf vf[4];
#pragma unroll
    for (int nt = 0; nt < 4; ++nt) {
      const __bf16* vp = vbase + (size_t)(nt * 16 + l15) * SEQ + j + lh * 16;
      vf[nt] = *(const v16bf*)vp;
    }

    // scale 1/sqrt(Dh); causal mask only on partial blocks (j+31 > q0)
#pragma unroll
    for (int half = 0; half < 2; ++half)
#pragma unroll
      for (int i = 0; i < 8; ++i) sc[half][i] *= 0.125f;
    if (causal && (j + 31 > q0)) {
#pragma unroll
      for (int half = 0; half < 2; ++half)
#pragma unroll
        for (int i = 0; i < 8; ++i) {
          const int qrow = q0 + i + 8 * lh;
          const int key  = j + half * 16 + l15;
          if (key > qrow) sc[half][i] = -1e30f;
        }
    }

    // online softmax; row i lives in element i, replicated over 16 lanes
#pragma unroll
    for (int i = 0; i < 8; ++i) {
      float mx = fmaxf(sc[0][i], sc[1][i]);
      mx = fmaxf(mx, __shfl_xor(mx, 1));
      mx = fmaxf(mx, __shfl_xor(mx, 2));
      mx = fmaxf(mx, __shfl_xor(mx, 4));
      mx = fmaxf(mx, __shfl_xor(mx, 8));
      const float mnew = fmaxf(m_run[i], mx);
      const float p0 = __expf(sc[0][i] - mnew);
      const float p1 = __expf(sc[1][i] - mnew);
      const float corr = __expf(m_run[i] - mnew);
      float ps = p0 + p1;
      ps += __shfl_xor(ps, 1);
      ps += __shfl_xor(ps, 2);
      ps += __shfl_xor(ps, 4);
      ps += __shfl_xor(ps, 8);
      l_run[i] = l_run[i] * corr + ps;
      m_run[i] = mnew;
#pragma unroll
      for (int nt = 0; nt < 4; ++nt) oacc[nt][i] *= corr;
      pbuf[wid][(i + 8 * lh) * 32 + l15]      = (__bf16)p0;
      pbuf[wid][(i + 8 * lh) * 32 + 16 + l15] = (__bf16)p1;
    }
    // repack P (16x32) from C-layout to A-layout (in-order DS ops, 1 wave)
    const __bf16* pp = &pbuf[wid][l15 * 32];
    v16bf pf = join8(*(const v8bf*)(pp + lh * 16),
                     *(const v8bf*)(pp + 8 + lh * 16));
    // O += P * V  (V fragments landed during softmax)
#pragma unroll
    for (int nt = 0; nt < 4; ++nt)
      oacc[nt] = wmma_bf16(pf, vf[nt], oacc[nt]);
  }

#pragma unroll
  for (int i = 0; i < 8; ++i) {
    const float inv = 1.0f / l_run[i];
    const int qrow = q0 + i + 8 * lh;
#pragma unroll
    for (int nt = 0; nt < 4; ++nt) {
      O[(size_t)(b * SEQ + qrow) * D_EMBED + h * D_HEAD + nt * 16 + l15] =
          (__bf16)(oacc[nt][i] * inv);
    }
  }
}

// ---------------------------------------------------------------- launch
extern "C" void kernel_launch(void* const* d_in, const int* in_sizes, int n_in,
                              void* d_out, int out_size, void* d_ws,
                              size_t ws_size, hipStream_t stream) {
  const float* x      = (const float*)d_in[0];   // [B,S,D] f32
  const float* wqkv   = (const float*)d_in[1];   // [3D,D]  f32
  const float* bqkv   = (const float*)d_in[2];   // [3D]    f32
  const float* wout   = (const float*)d_in[3];   // [D,D]   f32
  const float* bout   = (const float*)d_in[4];   // [D]     f32
  const int*   causal = (const int*)d_in[5];     // scalar
  float* out = (float*)d_out;                    // [B,S,D] f32

  char* ws = (char*)d_ws;
  __bf16* xbf   = (__bf16*)(ws);                 // 4,194,304  (8 MiB)
  __bf16* wqkvb = (__bf16*)(ws + 8388608);       // 3,145,728  (6 MiB)
  __bf16* woutb = (__bf16*)(ws + 14680064);      // 1,048,576  (2 MiB)
  __bf16* qkvb  = (__bf16*)(ws + 16777216);      // 12,582,912 (24 MiB)
  __bf16* vtb   = (__bf16*)(ws + 41943040);      // 4,194,304  (8 MiB)
  __bf16* abf   = (__bf16*)(ws + 50331648);      // 4,194,304  (8 MiB)

  const int n_x  = BS * D_EMBED;                 // 4,194,304
  const int n_wq = QKV_W * D_EMBED;              // 3,145,728
  const int n_wo = D_EMBED * D_EMBED;            // 1,048,576
  const int n_vt = BATCH * N_HEADS * D_HEAD * SEQ;  // 4,194,304

  cvt_f32_bf16_kernel<<<(n_x + 255) / 256, 256, 0, stream>>>(x, xbf, n_x);
  cvt_f32_bf16_kernel<<<(n_wq + 255) / 256, 256, 0, stream>>>(wqkv, wqkvb, n_wq);
  cvt_f32_bf16_kernel<<<(n_wo + 255) / 256, 256, 0, stream>>>(wout, woutb, n_wo);

  // QKV projection: [4096,3072] = xbf[4096,1024] * wqkvb^T + bqkv
  {
    dim3 grid(QKV_W / 256, BS / 64);
    gemm_bf16_wmma_kernel<__bf16><<<grid, 256, 0, stream>>>(
        xbf, wqkvb, bqkv, qkvb, BS, QKV_W, D_EMBED);
  }

  build_vt_kernel<<<(n_vt + 255) / 256, 256, 0, stream>>>(qkvb, vtb);

  // flash attention: 4096 waves total, 8 waves per block
  flash_attn_kernel<<<512, 256, 0, stream>>>(qkvb, vtb, abf, causal);

  // out projection: [4096,1024] f32 = abf * woutb^T + bout
  {
    dim3 grid(D_EMBED / 256, BS / 64);
    gemm_bf16_wmma_kernel<float><<<grid, 256, 0, stream>>>(
        abf, woutb, bout, out, BS, D_EMBED, D_EMBED);
  }
}
